// EEG_CNN_xLSTM_18983755448857
// MI455X (gfx1250) — matrix operand order repaired and checked by reading, not compiled, not run on hardware
//
#include <hip/hip_runtime.h>
#include <math.h>

// ---------------------------------------------------------------------------
// MI455X (gfx1250) implementation: bf16 WMMA GEMMs for conv1/conv2/res/pre,
// persistent wave32 WMMA kernel for the sLSTM recurrence, fused BN/ReLU
// epilogues. f32 accumulation everywhere; gate nonlinearity in f32.
// This revision: load/compute decoupling inside each k-step — all B
// fragments for a step are fetched into distinct registers (single s_clause
// burst), then the WMMA chain issues back-to-back without per-tile waits.
// ---------------------------------------------------------------------------

typedef __attribute__((ext_vector_type(16))) __bf16 v16bf;
typedef __attribute__((ext_vector_type(8)))  __bf16 v8bf;
typedef __attribute__((ext_vector_type(8)))  float  v8f;

#define WMMA_BF16(A,B,C) \
  __builtin_amdgcn_wmma_f32_16x16x32_bf16(false,(A),false,(B),(short)0,(C),false,false)

constexpr int Bn  = 64;
constexpr int T   = 4096;
constexpr int CIN = 64;
constexpr int K1  = 128;
constexpr int Hd  = 96;
constexpr int TP  = 1024;   // T/4
constexpr int G4  = 384;    // 4*H
constexpr int NC  = 109;

// ---------------- helper: gather one 16x32 bf16 A fragment ------------------
// Documented layout: lane L holds row M=L%16; VGPR0-3 = K = grp*8 + 0..7,
// VGPR4-7 = K = 16 + grp*8 + 0..7  (grp = L/16).  Two contiguous 16B runs.
__device__ __forceinline__ v16bf gather_a(const __bf16* __restrict__ row,
                                          int c0, bool valid) {
  v16bf a;
  if (valid) {
    v8bf lo = *(const v8bf*)(row + c0);
    v8bf hi = *(const v8bf*)(row + c0 + 16);
#pragma unroll
    for (int i = 0; i < 8; ++i) { a[i] = lo[i]; a[8 + i] = hi[i]; }
  } else {
#pragma unroll
    for (int i = 0; i < 16; ++i) a[i] = (__bf16)0.0f;
  }
  return a;
}

__device__ __forceinline__ v16bf load_bfrag(const __bf16* __restrict__ base,
                                            int f, int lane) {
  return *(const v16bf*)(base + ((size_t)f * 32 + lane) * 16);
}

// ---------------- tiny prep kernels ----------------------------------------
__global__ void cast_kernel(const float* __restrict__ in,
                            __bf16* __restrict__ out, int n) {
  int i = blockIdx.x * 256 + threadIdx.x;
  if (i < n) out[i] = (__bf16)in[i];
}

// Fold BN (eval) + conv bias:  y*s + t, s = g/sqrt(v+eps), t = (cb-m)*s + b
__global__ void affine_kernel(const float* __restrict__ g, const float* __restrict__ b,
                              const float* __restrict__ m, const float* __restrict__ v,
                              const float* __restrict__ cb,
                              float* __restrict__ s, float* __restrict__ t, int n) {
  int i = blockIdx.x * blockDim.x + threadIdx.x;
  if (i < n) {
    float sv = g[i] * rsqrtf(v[i] + 1e-5f);
    s[i] = sv;
    t[i] = (cb[i] - m[i]) * sv + b[i];
  }
}

// Pack weights as WMMA B fragments (32x16 bf16 per frag, frag f = kt*NT+nt).
// B layout: lane = N col; element e of the 16 halves -> K offset =
// grp*8 + e for e<8, 16 + grp*8 + (e-8) for e>=8 (matches A fragment).
// mode 0: conv1 w [128][64][7],  k = dk*64+c
// mode 1: conv2 w [96][128][3],  k = dk*128+c
// mode 2: res   w [96][128][1],  k = c
// mode 3: wx/wh  [384][96],      k = h_in
__global__ void pack_kernel(const float* __restrict__ src, __bf16* __restrict__ dst,
                            int NT, int mode, int total) {
  int gid = blockIdx.x * blockDim.x + threadIdx.x;
  if (gid >= total) return;
  int lane = gid & 31;
  int f    = gid >> 5;
  int kt   = f / NT, nt = f % NT;
  int kgrp = (lane >> 4) * 8;
  int n    = nt * 16 + (lane & 15);
#pragma unroll
  for (int e = 0; e < 16; ++e) {
    int k = kt * 32 + kgrp + (e & 7) + (e >> 3) * 16;
    size_t idx;
    if (mode == 0)      idx = (size_t)n * 448 + (size_t)(k & 63)  * 7 + (k >> 6);
    else if (mode == 1) idx = (size_t)n * 384 + (size_t)(k & 127) * 3 + (k >> 7);
    else if (mode == 2) idx = (size_t)n * 128 + k;
    else                idx = (size_t)n * 96  + k;
    dst[((size_t)f * 32 + lane) * 16 + e] = (__bf16)src[idx];
  }
}

// ---------------- conv1 (+BN1+ReLU) as implicit GEMM -----------------------
// M tile 64 (4 mtiles), N = 128 (8 ntiles). Wave w: mtile=w>>1, 4 ntiles.
__global__ void __launch_bounds__(256) conv1_kernel(
    const __bf16* __restrict__ xbf, const __bf16* __restrict__ w1p,
    const float* __restrict__ s1, const float* __restrict__ t1,
    __bf16* __restrict__ y1) {
  int tid = threadIdx.x, lane = tid & 31, wave = tid >> 5;
  int grp = lane >> 4;
  int mt = wave >> 1, nq = wave & 1;
  size_t rowbase = (size_t)blockIdx.x * 64 + mt * 16;
  size_t r = rowbase + (lane & 15);
  int b = (int)(r >> 12);
  int t = (int)(r & 4095);

  v8f acc[4];
#pragma unroll
  for (int i = 0; i < 4; ++i)
#pragma unroll
    for (int j = 0; j < 8; ++j) acc[i][j] = 0.f;

  for (int kt = 0; kt < 14; ++kt) {
    int dk = kt >> 1;
    int cbase = (kt & 1) * 32;
    int tt = t + dk - 3;
    const __bf16* row = xbf + ((size_t)b * T + tt) * CIN;
    // ---- load phase: A + all 4 B fragments into live registers ----
    v16bf a = gather_a(row, cbase + grp * 8, (tt >= 0) && (tt < T));
    v16bf bb[4];
#pragma unroll
    for (int i = 0; i < 4; ++i)
      bb[i] = load_bfrag(w1p, kt * 8 + (nq * 4 + i), lane);
    // ---- compute phase: back-to-back WMMAs ----
#pragma unroll
    for (int i = 0; i < 4; ++i)
      acc[i] = WMMA_BF16(a, bb[i], acc[i]);
  }
#pragma unroll
  for (int i = 0; i < 4; ++i) {
    int nn = (nq * 4 + i) * 16 + (lane & 15);
    float sc = s1[nn], sh = t1[nn];
#pragma unroll
    for (int j = 0; j < 8; ++j) {
      size_t rr = rowbase + j + grp * 8;   // C/D layout: M = j + 8*grp
      float v = fmaxf(acc[i][j] * sc + sh, 0.f);
      y1[rr * K1 + nn] = (__bf16)v;
    }
  }
}

// ---------------- conv2(+BN2+ReLU) + res(+BNr), fused ----------------------
// M tile 64, N = 96 (6 ntiles); wave: mtile=w>>1, 3 ntiles; two acc sets.
__global__ void __launch_bounds__(256) conv2_kernel(
    const __bf16* __restrict__ y1, const __bf16* __restrict__ w2p,
    const __bf16* __restrict__ wrp,
    const float* __restrict__ s2, const float* __restrict__ t2,
    const float* __restrict__ sr, const float* __restrict__ tr,
    __bf16* __restrict__ seq) {
  int tid = threadIdx.x, lane = tid & 31, wave = tid >> 5;
  int grp = lane >> 4;
  int mt = wave >> 1, ng = wave & 1;
  size_t rowbase = (size_t)blockIdx.x * 64 + mt * 16;
  size_t r = rowbase + (lane & 15);
  int b = (int)(r >> 10);
  int tp = (int)(r & 1023);

  v8f acc2[3], accr[3];
#pragma unroll
  for (int i = 0; i < 3; ++i)
#pragma unroll
    for (int j = 0; j < 8; ++j) { acc2[i][j] = 0.f; accr[i][j] = 0.f; }

  // conv2: K = 3*128, k = dk*128 + c
  for (int kt = 0; kt < 12; ++kt) {
    int dk = kt >> 2;
    int cbase = (kt & 3) * 32;
    int tt = 4 * tp + dk - 1;
    const __bf16* row = y1 + ((size_t)b * T + tt) * K1;
    v16bf a = gather_a(row, cbase + grp * 8, (tt >= 0) && (tt < T));
    v16bf bb[3];
#pragma unroll
    for (int i = 0; i < 3; ++i)
      bb[i] = load_bfrag(w2p, kt * 6 + (ng * 3 + i), lane);
#pragma unroll
    for (int i = 0; i < 3; ++i)
      acc2[i] = WMMA_BF16(a, bb[i], acc2[i]);
  }
  // residual: K = 128, position 4*tp (always in range)
  for (int kt = 0; kt < 4; ++kt) {
    const __bf16* row = y1 + ((size_t)b * T + 4 * tp) * K1;
    v16bf a = gather_a(row, kt * 32 + grp * 8, true);
    v16bf bb[3];
#pragma unroll
    for (int i = 0; i < 3; ++i)
      bb[i] = load_bfrag(wrp, kt * 6 + (ng * 3 + i), lane);
#pragma unroll
    for (int i = 0; i < 3; ++i)
      accr[i] = WMMA_BF16(a, bb[i], accr[i]);
  }
#pragma unroll
  for (int i = 0; i < 3; ++i) {
    int nn = (ng * 3 + i) * 16 + (lane & 15);
    float sc2 = s2[nn], sh2 = t2[nn], scr = sr[nn], shr = tr[nn];
#pragma unroll
    for (int j = 0; j < 8; ++j) {
      size_t rr = rowbase + j + grp * 8;
      float v = fmaxf(acc2[i][j] * sc2 + sh2, 0.f) + (accr[i][j] * scr + shr);
      seq[rr * Hd + nn] = (__bf16)v;
    }
  }
}

// ---------------- pre = seq @ wx^T + bias  (M=65536, K=96, N=384) ----------
// grid.y halves N; per WG: M=64, N=192 (12 ntiles); wave: mtile=w>>1, 6 ntiles.
__global__ void __launch_bounds__(256) pre_kernel(
    const __bf16* __restrict__ seq, const __bf16* __restrict__ wxp,
    const float* __restrict__ bias, float* __restrict__ pre) {
  int tid = threadIdx.x, lane = tid & 31, wave = tid >> 5;
  int grp = lane >> 4;
  int mt = wave >> 1, ng = wave & 1;
  int nhalf = blockIdx.y;
  size_t rowbase = (size_t)blockIdx.x * 64 + mt * 16;
  size_t r = rowbase + (lane & 15);

  v8f acc[6];
#pragma unroll
  for (int i = 0; i < 6; ++i)
#pragma unroll
    for (int j = 0; j < 8; ++j) acc[i][j] = 0.f;

  for (int kt = 0; kt < 3; ++kt) {
    const __bf16* row = seq + r * Hd;
    v16bf a = gather_a(row, kt * 32 + grp * 8, true);
    v16bf bb[6];
#pragma unroll
    for (int i = 0; i < 6; ++i)
      bb[i] = load_bfrag(wxp, kt * 24 + (nhalf * 12 + ng * 6 + i), lane);
#pragma unroll
    for (int i = 0; i < 6; ++i)
      acc[i] = WMMA_BF16(a, bb[i], acc[i]);
  }
#pragma unroll
  for (int i = 0; i < 6; ++i) {
    int nn = (nhalf * 12 + ng * 6 + i) * 16 + (lane & 15);
    float bv = bias[nn];
#pragma unroll
    for (int j = 0; j < 8; ++j) {
      size_t rr = rowbase + j + grp * 8;
      pre[rr * G4 + nn] = acc[i][j] + bv;
    }
  }
}

// ---------------- persistent sLSTM recurrence ------------------------------
// 4 WGs x 16 batch rows. wh fragments resident in registers (9 per wave),
// h in LDS (A layout source), pre-activations prefetched into registers
// before the step barrier so their global latency overlaps the WMMA phase,
// state (c, n, m_stab, pooled) in thread registers. 1024 sequential steps.
__global__ void __launch_bounds__(256) slstm_kernel(
    const float* __restrict__ pre, const __bf16* __restrict__ whp,
    float* __restrict__ pooled) {
  __shared__ __align__(16) __bf16 h_lds[16 * Hd];
  __shared__ float g_lds[16 * G4];
  int tid = threadIdx.x, lane = tid & 31, wave = tid >> 5;
  int grp = lane >> 4;
  int mrow = lane & 15;
  int bbase = blockIdx.x * 16;

  v16bf wh[3][3];
#pragma unroll
  for (int nt = 0; nt < 3; ++nt)
#pragma unroll
    for (int kt = 0; kt < 3; ++kt)
      wh[nt][kt] = load_bfrag(whp, kt * 24 + (wave * 3 + nt), lane);

  for (int i = tid; i < 16 * Hd; i += 256) h_lds[i] = (__bf16)0.0f;

  // per-thread gate element mapping (fixed across steps)
  int eb[6], eh[6];
  const float* prow[6];
#pragma unroll
  for (int j = 0; j < 6; ++j) {
    int idx = tid + 256 * j;               // 1536 = 16*96 elements
    eb[j] = idx / Hd;
    eh[j] = idx % Hd;
    prow[j] = pre + (size_t)(bbase + eb[j]) * TP * G4 + eh[j];
  }

  float cc[6], nst[6], ms[6], pool[6];
#pragma unroll
  for (int j = 0; j < 6; ++j) { cc[j] = 0.f; nst[j] = 0.f; ms[j] = 0.f; pool[j] = 0.f; }

  for (int t = 0; t < TP; ++t) {
    // -------- prefetch this step's pre-activations into registers -------
    // Issued BEFORE the barrier: global latency overlaps the WMMA phase.
    float pz[6], pi[6], pf[6], po[6];
#pragma unroll
    for (int j = 0; j < 6; ++j) {
      const float* p = prow[j] + (size_t)t * G4;
      pz[j] = p[0];
      pi[j] = p[96];
      pf[j] = p[192];
      po[j] = p[288];
    }
    if (t + 1 < TP) {
#pragma unroll
      for (int j = 0; j < 6; ++j)
        __builtin_prefetch(prow[j] + (size_t)(t + 1) * G4, 0, 1);
    }
    __syncthreads();                       // h_lds ready
    // -------- GEMM phase: g_raw = h @ wh^T ------------------------------
    v16bf a[3];
#pragma unroll
    for (int kt = 0; kt < 3; ++kt)
      a[kt] = gather_a(h_lds + mrow * Hd, kt * 32 + grp * 8, true);
#pragma unroll
    for (int nt = 0; nt < 3; ++nt) {
      v8f acc;
#pragma unroll
      for (int j = 0; j < 8; ++j) acc[j] = 0.f;
#pragma unroll
      for (int kt = 0; kt < 3; ++kt)
        acc = WMMA_BF16(a[kt], wh[nt][kt], acc);
      int nn = (wave * 3 + nt) * 16 + (lane & 15);
#pragma unroll
      for (int j = 0; j < 8; ++j) g_lds[(j + grp * 8) * G4 + nn] = acc[j];
    }
    __syncthreads();                       // g_lds ready
    // -------- gate phase (exp-stabilized sLSTM, f32) --------------------
#pragma unroll
    for (int j = 0; j < 6; ++j) {
      int gbase = eb[j] * G4 + eh[j];
      float zp = g_lds[gbase]       + pz[j];
      float ip = g_lds[gbase + 96]  + pi[j];
      float fp = g_lds[gbase + 192] + pf[j];
      float op = g_lds[gbase + 288] + po[j];
      float zv = tanhf(zp);
      float ov = 1.f / (1.f + expf(-op));
      float mn = fmaxf(fp + ms[j], ip);
      float iv = expf(ip - mn);
      float fv = expf(fp + ms[j] - mn);
      cc[j]  = fv * cc[j] + iv * zv;
      nst[j] = fv * nst[j] + iv;
      float hv = ov * (cc[j] / fmaxf(nst[j], 1.0f));
      ms[j] = mn;
      pool[j] += hv;
      h_lds[eb[j] * Hd + eh[j]] = (__bf16)hv;
    }
  }
#pragma unroll
  for (int j = 0; j < 6; ++j) {
    pooled[(size_t)(bbase + eb[j]) * Hd + eh[j]] = pool[j] * (1.0f / TP);
  }
}

// ---------------- head: fc1 + ReLU + fc2 (tiny, VALU) ----------------------
__global__ void __launch_bounds__(256) head_kernel(
    const float* __restrict__ pooled,
    const float* __restrict__ f1w, const float* __restrict__ f1b,
    const float* __restrict__ f2w, const float* __restrict__ f2b,
    float* __restrict__ out) {
  __shared__ float z1[64 * 128];
  int tid = threadIdx.x;
  for (int idx = tid; idx < 64 * 128; idx += 256) {
    int b = idx >> 7, o = idx & 127;
    float a = f1b[o];
    for (int h = 0; h < Hd; ++h) a += pooled[b * Hd + h] * f1w[o * Hd + h];
    z1[idx] = fmaxf(a, 0.f);
  }
  __syncthreads();
  for (int idx = tid; idx < 64 * NC; idx += 256) {
    int b = idx / NC, o = idx % NC;
    float a = f2b[o];
    for (int k = 0; k < 128; ++k) a += z1[b * 128 + k] * f2w[o * 128 + k];
    out[(size_t)b * NC + o] = a;
  }
}

// ---------------------------------------------------------------------------
extern "C" void kernel_launch(void* const* d_in, const int* in_sizes, int n_in,
                              void* d_out, int out_size, void* d_ws, size_t ws_size,
                              hipStream_t stream) {
  const float* x    = (const float*)d_in[0];
  const float* c1w  = (const float*)d_in[1];
  const float* c1b  = (const float*)d_in[2];
  const float* bn1g = (const float*)d_in[3];
  const float* bn1b = (const float*)d_in[4];
  const float* bn1m = (const float*)d_in[5];
  const float* bn1v = (const float*)d_in[6];
  const float* c2w  = (const float*)d_in[7];
  const float* c2b  = (const float*)d_in[8];
  const float* bn2g = (const float*)d_in[9];
  const float* bn2b = (const float*)d_in[10];
  const float* bn2m = (const float*)d_in[11];
  const float* bn2v = (const float*)d_in[12];
  const float* rw   = (const float*)d_in[13];
  const float* rb   = (const float*)d_in[14];
  const float* bnrg = (const float*)d_in[15];
  const float* bnrb = (const float*)d_in[16];
  const float* bnrm = (const float*)d_in[17];
  const float* bnrv = (const float*)d_in[18];
  const float* wx   = (const float*)d_in[19];
  const float* whw  = (const float*)d_in[20];
  const float* sb   = (const float*)d_in[21];
  const float* f1w  = (const float*)d_in[22];
  const float* f1b  = (const float*)d_in[23];
  const float* f2w  = (const float*)d_in[24];
  const float* f2b  = (const float*)d_in[25];
  float* out = (float*)d_out;

  // ---- workspace bump allocation (total ~215 MB) ----
  char* p = (char*)d_ws;
  auto alloc = [&](size_t bytes) -> char* {
    char* q = p;
    p += (bytes + 255) & ~(size_t)255;
    return q;
  };
  __bf16* xbf = (__bf16*)alloc((size_t)Bn * T * CIN * 2);
  __bf16* y1  = (__bf16*)alloc((size_t)Bn * T * K1 * 2);
  __bf16* seq = (__bf16*)alloc((size_t)Bn * TP * Hd * 2);
  float*  pre = (float*) alloc((size_t)Bn * TP * G4 * 4);
  __bf16* w1p = (__bf16*)alloc((size_t)448 * 128 * 2);
  __bf16* w2p = (__bf16*)alloc((size_t)384 * 96 * 2);
  __bf16* wrp = (__bf16*)alloc((size_t)128 * 96 * 2);
  __bf16* wxp = (__bf16*)alloc((size_t)96 * 384 * 2);
  __bf16* whp = (__bf16*)alloc((size_t)96 * 384 * 2);
  float* s1 = (float*)alloc(128 * 4);
  float* t1 = (float*)alloc(128 * 4);
  float* s2 = (float*)alloc(96 * 4);
  float* t2 = (float*)alloc(96 * 4);
  float* sr = (float*)alloc(96 * 4);
  float* tr = (float*)alloc(96 * 4);
  float* pooled = (float*)alloc(64 * 96 * 4);

  // ---- prep ----
  cast_kernel<<<(Bn * T * CIN + 255) / 256, 256, 0, stream>>>(x, xbf, Bn * T * CIN);
  affine_kernel<<<1, 128, 0, stream>>>(bn1g, bn1b, bn1m, bn1v, c1b, s1, t1, 128);
  affine_kernel<<<1, 96, 0, stream>>>(bn2g, bn2b, bn2m, bn2v, c2b, s2, t2, 96);
  affine_kernel<<<1, 96, 0, stream>>>(bnrg, bnrb, bnrm, bnrv, rb, sr, tr, 96);
  pack_kernel<<<14, 256, 0, stream>>>(c1w, w1p, 8, 0, 14 * 8 * 32);
  pack_kernel<<<9, 256, 0, stream>>>(c2w, w2p, 6, 1, 12 * 6 * 32);
  pack_kernel<<<3, 256, 0, stream>>>(rw, wrp, 6, 2, 4 * 6 * 32);
  pack_kernel<<<9, 256, 0, stream>>>(wx, wxp, 24, 3, 3 * 24 * 32);
  pack_kernel<<<9, 256, 0, stream>>>(whw, whp, 24, 3, 3 * 24 * 32);

  // ---- pipeline ----
  conv1_kernel<<<Bn * T / 64, 256, 0, stream>>>(xbf, w1p, s1, t1, y1);
  conv2_kernel<<<Bn * TP / 64, 256, 0, stream>>>(y1, w2p, wrp, s2, t2, sr, tr, seq);
  pre_kernel<<<dim3(Bn * TP / 64, 2), 256, 0, stream>>>(seq, wxp, sb, pre);
  slstm_kernel<<<4, 256, 0, stream>>>(pre, whp, pooled);
  head_kernel<<<1, 256, 0, stream>>>(pooled, f1w, f1b, f2w, f2b, out);
}
